// LocalAttention_35880156790794
// MI455X (gfx1250) — matrix operand order repaired and checked by reading, not compiled, not run on hardware
//
#include <hip/hip_runtime.h>
#include <math.h>

typedef __attribute__((ext_vector_type(16))) _Float16 v16h;
typedef __attribute__((ext_vector_type(8)))  _Float16 v8h;
typedef __attribute__((ext_vector_type(8)))  float    v8f;

#define BH_N     16      // B*H
#define HNUM     8
#define SEQ_T    8192
#define E_DIM    64
#define BSZ      128     // bucket size
#define NBUCKETS 64
#define WIN      256     // look_backward window (2 buckets)
#define SCALE    0.125f  // 64^-0.5
#define BIGNEG   3.402823466e38f

// One block = one (bh, bucket) tile. 256 threads = 8 waves (wave32).
// Wave w computes query rows [16w, 16w+16).
__global__ __launch_bounds__(256) void local_attn_wmma_kernel(
    const float* __restrict__ q, const float* __restrict__ k,
    const float* __restrict__ v, const float* __restrict__ rel,
    float* __restrict__ out)
{
  __shared__ _Float16 ks[WIN * E_DIM];   // 32KB  normalized key window, row-major
  __shared__ _Float16 vt[E_DIM * WIN];   // 32KB  V^T (E x WIN)
  __shared__ _Float16 emb_p[BSZ * WIN];  // 64KB  phase1: emb, phase3: P (aliased)
  __shared__ float    rsum[BSZ];

  const int bh     = blockIdx.x / NBUCKETS;
  const int bucket = blockIdx.x % NBUCKETS;
  const int h      = bh % HNUM;
  const int tid    = threadIdx.x;
  const int lane   = tid & 31;
  const int wave   = tid >> 5;          // 0..7
  const int hi     = (lane >> 4) & 1;   // half-group
  const int colL   = lane & 15;

  const size_t rowbase = (size_t)bh * SEQ_T + (size_t)bucket * BSZ;   // q/out row base
  const long   kwin0   = (long)bucket * BSZ - BSZ;                    // first key pos in window

  // ---------------- Phase 0: stage normalized K window + V^T (1 row / thread)
  {
    long kp = kwin0 + tid;
    if (kp >= 0) {
      const float* kr = k + ((size_t)bh * SEQ_T + (size_t)kp) * E_DIM;
      const float* vr = v + ((size_t)bh * SEQ_T + (size_t)kp) * E_DIM;
      float ss = 0.f;
      #pragma unroll
      for (int e = 0; e < E_DIM; e += 4) {
        float4 f = *(const float4*)(kr + e);
        ss += f.x*f.x + f.y*f.y + f.z*f.z + f.w*f.w;
      }
      float rn = 1.0f / fmaxf(sqrtf(ss), 1e-12f);
      #pragma unroll
      for (int e = 0; e < E_DIM; e += 4) {
        float4 f = *(const float4*)(kr + e);
        float4 g = *(const float4*)(vr + e);
        ks[tid * E_DIM + e + 0] = (_Float16)(f.x * rn);
        ks[tid * E_DIM + e + 1] = (_Float16)(f.y * rn);
        ks[tid * E_DIM + e + 2] = (_Float16)(f.z * rn);
        ks[tid * E_DIM + e + 3] = (_Float16)(f.w * rn);
        vt[(e + 0) * WIN + tid] = (_Float16)g.x;
        vt[(e + 1) * WIN + tid] = (_Float16)g.y;
        vt[(e + 2) * WIN + tid] = (_Float16)g.z;
        vt[(e + 3) * WIN + tid] = (_Float16)g.w;
      }
    } else {                                   // look-around padding (masked later)
      #pragma unroll
      for (int e = 0; e < E_DIM; e++) {
        ks[tid * E_DIM + e] = (_Float16)0.0f;
        vt[e * WIN + tid]   = (_Float16)(-1.0f);
      }
    }
  }

  // ---------------- Q A-fragments for this wave (held in VGPRs, reused twice)
  // A order used here (contiguous k, matching B): lane L -> row L%16; lanes<16: K {0-15}; lanes>=16: K {16-31}
  const int m0 = wave * 16;
  v16h a0, a1;
  {
    const float* qr = q + (rowbase + (size_t)(m0 + colL)) * E_DIM;
    const int ko = hi ? 16 : 0;
    #pragma unroll
    for (int e = 0; e < 8; e++) {
      a0[e]     = (_Float16)qr[ko + e];
      a0[e + 8] = (_Float16)qr[ko + 8 + e];
      a1[e]     = (_Float16)qr[32 + ko + e];
      a1[e + 8] = (_Float16)qr[32 + ko + 8 + e];
    }
  }

  // ---------------- Phase 1: emb = Q . Rel^T  (16x256 strip), f16 into LDS
  // B order used here (contiguous k, matching A): lane L -> column L%16, 16 contiguous K values (hi: K+16)
  // unroll 1: keeps global-load pipelining (and VGPR pressure) bounded.
  #pragma unroll 1
  for (int t = 0; t < 16; t++) {
    const int j = t * 16 + colL;                           // rel row index (B column)
    const float* rr0 = rel + ((size_t)j * HNUM + h) * E_DIM;
    const int ko = hi ? 16 : 0;
    v16h b0, b1;
    #pragma unroll
    for (int e = 0; e < 16; e++) {
      b0[e] = (_Float16)rr0[ko + e];
      b1[e] = (_Float16)rr0[32 + ko + e];
    }
    v8f c = {};
    c = __builtin_amdgcn_wmma_f32_16x16x32_f16(false, a0, false, b0, (short)0, c, false, false);
    c = __builtin_amdgcn_wmma_f32_16x16x32_f16(false, a1, false, b1, (short)0, c, false, false);
    #pragma unroll
    for (int r = 0; r < 8; r++)                            // D: lane holds (m0+r+8*hi, 16t+colL)
      emb_p[(m0 + r + 8 * hi) * WIN + t * 16 + colL] = (_Float16)c[r];
  }

  __syncthreads();   // ks / vt ready for all waves

  // ---------------- Phase 2: S strip = Q . Kwin^T  (16x256 in registers)
  // Must be fully unrolled: acc[] lives in VGPRs across the softmax epilogue.
  v8f acc[16];
  #pragma unroll
  for (int t = 0; t < 16; t++) {
    const _Float16* kb = &ks[(t * 16 + colL) * E_DIM + (hi ? 16 : 0)];
    v16h b0 = *(const v16h*)(kb);
    v16h b1 = *(const v16h*)(kb + 32);
    v8f c = {};
    c = __builtin_amdgcn_wmma_f32_16x16x32_f16(false, a0, false, b0, (short)0, c, false, false);
    c = __builtin_amdgcn_wmma_f32_16x16x32_f16(false, a1, false, b1, (short)0, c, false, false);
    acc[t] = c;
  }

  // ---------------- Phase 3: shifted bias + masks + softmax (row-strip in registers)
  // Row i's 256 values live in one 16-lane half-group -> 4-step shfl_xor reductions.
  #pragma unroll
  for (int r = 0; r < 8; r++) {
    const int i = m0 + r + 8 * hi;          // query row within bucket
    float mx = -BIGNEG;
    #pragma unroll
    for (int t = 0; t < 16; t++) {
      const int jj = t * 16 + colL;
      float s = acc[t][r] * SCALE;
      const int rr = jj + 127 - i;          // shifted rel index; rr>255 is causal-masked
      if (rr < 256) s += SCALE * (float)emb_p[i * WIN + rr];
      const int  dkq = jj - 128 - i;        // key_pos - query_pos
      const long kp  = kwin0 + jj;
      if (dkq == 0) s = -50000.0f;          // shared-QK self mask
      if (dkq > 0 || kp < 0) s = -BIGNEG;   // causal / pad mask
      acc[t][r] = s;
      mx = fmaxf(mx, s);
    }
    #pragma unroll
    for (int off = 1; off < 16; off <<= 1)
      mx = fmaxf(mx, __shfl_xor(mx, off, 32));
    float sum = 0.f;
    #pragma unroll
    for (int t = 0; t < 16; t++) {
      const int jj = t * 16 + colL;
      float p = __expf(acc[t][r] - mx);     // masked -> exactly 0
      sum += p;
      emb_p[i * WIN + jj] = (_Float16)p;    // overwrite own row: emb consumed above
    }
    #pragma unroll
    for (int off = 1; off < 16; off <<= 1)
      sum += __shfl_xor(sum, off, 32);
    if (colL == 0) rsum[i] = sum;
  }

  __syncthreads();

  // ---------------- Phase 4: O strip = P . Vwin  (16x64), normalize by 1/rowsum
  // unroll 1: only the 4 v8f accumulators persist across iterations.
  v8f o0 = {}, o1 = {}, o2 = {}, o3 = {};
  #pragma unroll 1
  for (int kb8 = 0; kb8 < 8; kb8++) {
    const int k0 = kb8 * 32;
    v16h a;                                           // A-frag of P from LDS
    {
      const _Float16* pr = &emb_p[(m0 + colL) * WIN + k0 + (hi ? 16 : 0)];
      v8h lo = *(const v8h*)(pr);
      v8h hh = *(const v8h*)(pr + 8);
      #pragma unroll
      for (int e = 0; e < 8; e++) { a[e] = lo[e]; a[e + 8] = hh[e]; }
    }
    const int vo = k0 + (hi ? 16 : 0);
    v16h b0 = *(const v16h*)&vt[(0 * 16 + colL) * WIN + vo];
    v16h b1 = *(const v16h*)&vt[(1 * 16 + colL) * WIN + vo];
    v16h b2 = *(const v16h*)&vt[(2 * 16 + colL) * WIN + vo];
    v16h b3 = *(const v16h*)&vt[(3 * 16 + colL) * WIN + vo];
    o0 = __builtin_amdgcn_wmma_f32_16x16x32_f16(false, a, false, b0, (short)0, o0, false, false);
    o1 = __builtin_amdgcn_wmma_f32_16x16x32_f16(false, a, false, b1, (short)0, o1, false, false);
    o2 = __builtin_amdgcn_wmma_f32_16x16x32_f16(false, a, false, b2, (short)0, o2, false, false);
    o3 = __builtin_amdgcn_wmma_f32_16x16x32_f16(false, a, false, b3, (short)0, o3, false, false);
  }

  #pragma unroll
  for (int r = 0; r < 8; r++) {
    const int m = m0 + r + 8 * hi;
    const float inv = 1.0f / rsum[m];
    float* orow = out + (rowbase + (size_t)m) * E_DIM;
    orow[ 0 + colL] = o0[r] * inv;
    orow[16 + colL] = o1[r] * inv;
    orow[32 + colL] = o2[r] * inv;
    orow[48 + colL] = o3[r] * inv;
  }
}

extern "C" void kernel_launch(void* const* d_in, const int* in_sizes, int n_in,
                              void* d_out, int out_size, void* d_ws, size_t ws_size,
                              hipStream_t stream) {
  (void)in_sizes; (void)n_in; (void)d_ws; (void)ws_size; (void)out_size;
  const float* q   = (const float*)d_in[0];
  const float* k   = (const float*)d_in[1];
  const float* v   = (const float*)d_in[2];
  const float* rel = (const float*)d_in[3];
  float* out = (float*)d_out;
  dim3 grid(BH_N * NBUCKETS);   // 1024 tiles
  dim3 block(256);              // 8 wave32
  local_attn_wmma_kernel<<<grid, block, 0, stream>>>(q, k, v, rel, out);
}
